// SelfAttentionV0_71322226917692
// MI455X (gfx1250) — matrix operand order repaired and checked by reading, not compiled, not run on hardware
//
#include <hip/hip_runtime.h>
#include <hip/hip_bf16.h>
#include <math.h>

#define BATCH  2
#define SEQ    2048
#define DMODEL 1024
#define NHEAD  16
#define HDIM   64
#define MTOT   (BATCH * SEQ)   // 4096

typedef __attribute__((ext_vector_type(16))) __bf16 v16bf;
typedef __attribute__((ext_vector_type(8)))  float  v8f;

struct __align__(8) bf16x4 { __bf16 a, b, c, d; };

// ---------------------------------------------------------------------------
// WMMA helpers (CDNA5: V_WMMA_F32_16X16X32_BF16, wave32)
// ---------------------------------------------------------------------------
__device__ __forceinline__ v8f wmma_bf16(v16bf a, v16bf b, v8f c) {
  // 8 args: (neg_a, A, neg_b, B, c_mod, C, reuse_a, reuse_b)
  return __builtin_amdgcn_wmma_f32_16x16x32_bf16(false, a, false, b,
                                                 (short)0, c, false, false);
}

// A fragment: 16x32 (MxK) bf16, row-major source, leading dim ld (elements).
// Layout: lanes 0-15 -> M=lane, K=0..15;
//         lanes 16-31 -> M=lane-16, K=16..31.
__device__ __forceinline__ v16bf load_fragA(const __bf16* p, int ld) {
  const int lane = threadIdx.x & 31;
  const __bf16* row = p + (lane & 15) * ld + ((lane >> 4) << 4);
  v16bf f;
#pragma unroll
  for (int i = 0; i < 16; ++i) f[i] = row[i];
  return f;
}

// B fragment: 32x16 (KxN) bf16, source stored N-major (bt[n][k]), lead dim ld.
// Layout used here (contiguous k, same as A): lanes 0-15 -> N=lane, K=0..15;
//                                   lanes 16-31 -> N=lane-16, K=16..31.
__device__ __forceinline__ v16bf load_fragB(const __bf16* bt, int ld) {
  const int lane = threadIdx.x & 31;
  const __bf16* row = bt + (lane & 15) * ld + ((lane >> 4) << 4);
  v16bf f;
#pragma unroll
  for (int i = 0; i < 16; ++i) f[i] = row[i];
  return f;
}

// ---------------------------------------------------------------------------
// CDNA5 async global->LDS DMA (ASYNCcnt-tracked), GVS addressing mode:
//   mem_addr = SADDR(64) + VOFFSET(32) + IOFFSET ; LDS dest byte addr in VDST
// ---------------------------------------------------------------------------
__device__ __forceinline__ void async_ld128(unsigned lds_byte_addr,
                                            unsigned vbyte_off,
                                            const void* sbase) {
  asm volatile("global_load_async_to_lds_b128 %0, %1, %2"
               :
               : "v"(lds_byte_addr), "v"(vbyte_off),
                 "s"((unsigned long long)(size_t)sbase)
               : "memory");
}
__device__ __forceinline__ void wait_asynccnt0() {
  asm volatile("s_wait_asynccnt 0x0" ::: "memory");
}

// C/D 16x16 f32 layout: element (m,n) for vgpr i: m = i + 8*(lane>>4), n = lane&15

// ---------------------------------------------------------------------------
// Kernel 1: fused QKV projection.  x[4096,1024] @ W[1024,1024] + bias.
// blockIdx.z selects {Q,K,V}.  Output Q/K: [b][h][s][d] bf16, V: [b][h][d][s].
// ---------------------------------------------------------------------------
__global__ __launch_bounds__(128) void qkv_kernel(
    const float* __restrict__ x,
    const float* __restrict__ wq, const float* __restrict__ bq,
    const float* __restrict__ wk, const float* __restrict__ bk,
    const float* __restrict__ wv, const float* __restrict__ bv,
    __bf16* __restrict__ Qh, __bf16* __restrict__ Kh, __bf16* __restrict__ Vt)
{
  __shared__ __align__(16) __bf16 ldsA[64][32];   // x tile, row-major [m][k]
  __shared__ __align__(16) __bf16 ldsBt[64][32];  // W tile, transposed [n][k]

  const int tid  = threadIdx.x;
  const int wave = tid >> 5;
  const int lane = tid & 31;
  const int z    = blockIdx.z;
  const float* w    = (z == 0) ? wq : (z == 1) ? wk : wv;
  const float* bias = (z == 0) ? bq : (z == 1) ? bk : bv;
  const int m0 = blockIdx.y * 64;
  const int n0 = blockIdx.x * 64;

  v8f acc[4];
#pragma unroll
  for (int t = 0; t < 4; ++t) acc[t] = v8f{0.f,0.f,0.f,0.f,0.f,0.f,0.f,0.f};

  for (int kk = 0; kk < DMODEL; kk += 32) {
    // ---- batch all global b128 loads first (no per-element waits) ----
    float4 av[4], bvv[4];
#pragma unroll
    for (int e = 0; e < 4; ++e) {                // A: 512 float4 total
      int li = tid + 128 * e;
      int r = li >> 3, c = li & 7;               // 8 float4 per 32-wide row
      av[e] = ((const float4*)(x + (size_t)(m0 + r) * DMODEL + kk))[c];
    }
#pragma unroll
    for (int e = 0; e < 4; ++e) {                // B: 512 float4 total
      int li = tid + 128 * e;
      int r = li >> 4, c = li & 15;              // 16 float4 per 64-wide row
      bvv[e] = ((const float4*)(w + (size_t)(kk + r) * DMODEL + n0))[c];
    }
    // ---- convert + store to LDS ----
#pragma unroll
    for (int e = 0; e < 4; ++e) {
      int li = tid + 128 * e;
      int r = li >> 3, c = (li & 7) * 4;
      bf16x4 t{(__bf16)av[e].x, (__bf16)av[e].y,
               (__bf16)av[e].z, (__bf16)av[e].w};
      *(bf16x4*)&ldsA[r][c] = t;                 // 8-byte ds_store
    }
#pragma unroll
    for (int e = 0; e < 4; ++e) {                // transpose W into [n][k]
      int li = tid + 128 * e;
      int r = li >> 4, c = (li & 15) * 4;
      ldsBt[c + 0][r] = (__bf16)bvv[e].x;
      ldsBt[c + 1][r] = (__bf16)bvv[e].y;
      ldsBt[c + 2][r] = (__bf16)bvv[e].z;
      ldsBt[c + 3][r] = (__bf16)bvv[e].w;
    }
    __syncthreads();
    v16bf af = load_fragA(&ldsA[wave * 16][0], 32);
#pragma unroll
    for (int t = 0; t < 4; ++t) {
      v16bf bf = load_fragB(&ldsBt[t * 16][0], 32);
      acc[t] = wmma_bf16(af, bf, acc[t]);
    }
    __syncthreads();
  }

  const int nn = lane & 15;
  const int mh = (lane >> 4) * 8;
#pragma unroll
  for (int t = 0; t < 4; ++t) {
    int n_g = n0 + t * 16 + nn;
    float bv_ = bias[n_g];
    int h = n_g >> 6;            // / HDIM
    int d = n_g & (HDIM - 1);
#pragma unroll
    for (int i = 0; i < 8; ++i) {
      int m_g = m0 + wave * 16 + mh + i;
      int bb  = m_g >> 11;       // / SEQ
      int s   = m_g & (SEQ - 1);
      float val = acc[t][i] + bv_;
      if (z == 2) {
        Vt[(size_t)((bb * NHEAD + h) * HDIM + d) * SEQ + s] = (__bf16)val;
      } else {
        __bf16* dst = (z == 0) ? Qh : Kh;
        dst[(size_t)((bb * NHEAD + h) * SEQ + s) * HDIM + d] = (__bf16)val;
      }
    }
  }
}

// ---------------------------------------------------------------------------
// Kernel 2: causal flash-attention per (b,h).  Block = 128 thr (4 waves),
// each wave owns 16 q-rows; 64 q-rows per block, k processed in 64-blocks.
// K/V tiles are staged with CDNA5 async global->LDS DMA.
// ---------------------------------------------------------------------------
__global__ __launch_bounds__(128) void attn_kernel(
    const __bf16* __restrict__ Qh, const __bf16* __restrict__ Kh,
    const __bf16* __restrict__ Vt, __bf16* __restrict__ O)
{
  __shared__ __align__(16) __bf16 ldsK[64][64];       // [k_local][d]  (N-major for QK^T)
  __shared__ __align__(16) __bf16 ldsV[64][64];       // [d][k_local]  (N-major for P.V)
  __shared__ __align__(16) __bf16 ldsP[4][16][64];    // per-wave P tile [m][k]

  const int tid  = threadIdx.x;
  const int wave = tid >> 5;
  const int lane = tid & 31;
  const int q0 = blockIdx.x * 64;
  const int h = blockIdx.y, b = blockIdx.z;

  const __bf16* Qp = Qh + (size_t)((b * NHEAD + h) * SEQ) * HDIM;
  const __bf16* Kp = Kh + (size_t)((b * NHEAD + h) * SEQ) * HDIM;
  const __bf16* Vp = Vt + (size_t)((b * NHEAD + h) * HDIM) * SEQ;

  // generic->LDS: low 32 bits of a generic LDS address are the LDS byte offset
  const unsigned ldsKbase = (unsigned)(size_t)&ldsK[0][0];
  const unsigned ldsVbase = (unsigned)(size_t)&ldsV[0][0];

  // Q fragments for this wave's 16 rows (d 0..31 and 32..63), loaded once.
  v16bf qf0 = load_fragA(Qp + (size_t)(q0 + wave * 16) * HDIM, HDIM);
  v16bf qf1 = load_fragA(Qp + (size_t)(q0 + wave * 16) * HDIM + 32, HDIM);

  v8f oacc[4];
#pragma unroll
  for (int t = 0; t < 4; ++t) oacc[t] = v8f{0.f,0.f,0.f,0.f,0.f,0.f,0.f,0.f};
  float rm[8], rl[8];
#pragma unroll
  for (int i = 0; i < 8; ++i) { rm[i] = -3.0e38f; rl[i] = 0.f; }

  const int nn = lane & 15;
  const int mh = (lane >> 4) * 8;
  const int nblocks = blockIdx.x + 1;   // causal: only k-blocks <= q-block

  for (int j = 0; j < nblocks; ++j) {
    const int k0 = j * 64;
    __syncthreads();   // previous-iteration consumers done before DMA lands
    {
      const void* kbase = (const void*)(Kp + (size_t)k0 * HDIM);  // contiguous 8KB
      const void* vbase = (const void*)(Vp + k0);                 // rows stride SEQ
#pragma unroll
      for (int e = 0; e < 4; ++e) {
        int idx = tid + 128 * e;                 // 512 x 16B chunks per tile
        unsigned off = (unsigned)idx * 16u;      // also the LDS offset (rows contiguous)
        async_ld128(ldsKbase + off, off, kbase);
        unsigned voff = (unsigned)((idx >> 3) * (SEQ * 2) + (idx & 7) * 16);
        async_ld128(ldsVbase + off, voff, vbase);
      }
      if (j + 1 < nblocks)   // prefetch next K tile into cache hierarchy
        __builtin_prefetch((const char*)(Kp + (size_t)(k0 + 64) * HDIM), 0, 1);
      wait_asynccnt0();
    }
    __syncthreads();

    // ---- logits S = (Q K^T) * 1/sqrt(HD), causal masked ----
    float pv[4][8];
    float bm[8];
#pragma unroll
    for (int i = 0; i < 8; ++i) bm[i] = -3.0e38f;
    v8f zero = v8f{0.f,0.f,0.f,0.f,0.f,0.f,0.f,0.f};
#pragma unroll
    for (int t = 0; t < 4; ++t) {
      v16bf kf0 = load_fragB(&ldsK[t * 16][0], 64);
      v16bf kf1 = load_fragB(&ldsK[t * 16][32], 64);
      v8f s = wmma_bf16(qf0, kf0, zero);
      s = wmma_bf16(qf1, kf1, s);
      int k_idx = k0 + t * 16 + nn;
#pragma unroll
      for (int i = 0; i < 8; ++i) {
        int q_idx = q0 + wave * 16 + mh + i;
        float v = s[i] * 0.125f;                 // 1/sqrt(64)
        v = (k_idx <= q_idx) ? v : -1.0e30f;
        pv[t][i] = v;
        bm[i] = fmaxf(bm[i], v);
      }
    }
    // row max across the 16 lanes holding one row
#pragma unroll
    for (int off = 1; off < 16; off <<= 1)
#pragma unroll
      for (int i = 0; i < 8; ++i)
        bm[i] = fmaxf(bm[i], __shfl_xor(bm[i], off, 16));

    float corr[8], ls[8];
#pragma unroll
    for (int i = 0; i < 8; ++i) {
      float mn = fmaxf(rm[i], bm[i]);
      corr[i] = __expf(rm[i] - mn);
      rm[i] = mn;
      rl[i] *= corr[i];
      ls[i] = 0.f;
    }
#pragma unroll
    for (int t = 0; t < 4; ++t) {
#pragma unroll
      for (int i = 0; i < 8; ++i) {
        float p = __expf(pv[t][i] - rm[i]);
        ls[i] += p;
        ldsP[wave][mh + i][t * 16 + nn] = (__bf16)p;
      }
#pragma unroll
      for (int i = 0; i < 8; ++i) oacc[t][i] *= corr[i];
    }
#pragma unroll
    for (int off = 1; off < 16; off <<= 1)
#pragma unroll
      for (int i = 0; i < 8; ++i)
        ls[i] += __shfl_xor(ls[i], off, 16);
#pragma unroll
    for (int i = 0; i < 8; ++i) rl[i] += ls[i];

    __syncthreads();   // P tile visible for re-fragmentation as A operand

    // ---- O += P @ V ----
    v16bf pf0 = load_fragA(&ldsP[wave][0][0], 64);
    v16bf pf1 = load_fragA(&ldsP[wave][0][32], 64);
#pragma unroll
    for (int t = 0; t < 4; ++t) {
      v16bf vf0 = load_fragB(&ldsV[t * 16][0], 64);
      v16bf vf1 = load_fragB(&ldsV[t * 16][32], 64);
      oacc[t] = wmma_bf16(pf0, vf0, oacc[t]);
      oacc[t] = wmma_bf16(pf1, vf1, oacc[t]);
    }
  }

  // epilogue: O[b][s][h*64 + d] = oacc / rowsum   (bf16 for output projection)
#pragma unroll
  for (int t = 0; t < 4; ++t) {
#pragma unroll
    for (int i = 0; i < 8; ++i) {
      int q_idx = q0 + wave * 16 + mh + i;
      float val = oacc[t][i] / rl[i];
      O[(size_t)(b * SEQ + q_idx) * DMODEL + h * HDIM + t * 16 + nn] =
          (__bf16)val;
    }
  }
}

// ---------------------------------------------------------------------------
// Kernel 3: output projection.  O[4096,1024](bf16) @ wo[1024,1024] + bo -> f32
// ---------------------------------------------------------------------------
__global__ __launch_bounds__(128) void out_proj_kernel(
    const __bf16* __restrict__ O, const float* __restrict__ wo,
    const float* __restrict__ bo, float* __restrict__ out)
{
  __shared__ __align__(16) __bf16 ldsA[64][32];
  __shared__ __align__(16) __bf16 ldsBt[64][32];

  const int tid  = threadIdx.x;
  const int wave = tid >> 5;
  const int lane = tid & 31;
  const int m0 = blockIdx.y * 64;
  const int n0 = blockIdx.x * 64;

  v8f acc[4];
#pragma unroll
  for (int t = 0; t < 4; ++t) acc[t] = v8f{0.f,0.f,0.f,0.f,0.f,0.f,0.f,0.f};

  for (int kk = 0; kk < DMODEL; kk += 32) {
    uint4 av[2];
    float4 bvv[4];
#pragma unroll
    for (int e = 0; e < 2; ++e) {                // A: 256 uint4 of bf16
      int idx = tid + 128 * e;
      int r = idx >> 2, c = idx & 3;             // 4 uint4 per 32-wide bf16 row
      av[e] = ((const uint4*)(O + (size_t)(m0 + r) * DMODEL + kk))[c];
    }
#pragma unroll
    for (int e = 0; e < 4; ++e) {
      int li = tid + 128 * e;
      int r = li >> 4, c = li & 15;
      bvv[e] = ((const float4*)(wo + (size_t)(kk + r) * DMODEL + n0))[c];
    }
#pragma unroll
    for (int e = 0; e < 2; ++e) {
      int idx = tid + 128 * e;
      int r = idx >> 2, c = idx & 3;
      ((uint4*)&ldsA[r][0])[c] = av[e];
    }
#pragma unroll
    for (int e = 0; e < 4; ++e) {
      int li = tid + 128 * e;
      int r = li >> 4, c = (li & 15) * 4;
      ldsBt[c + 0][r] = (__bf16)bvv[e].x;
      ldsBt[c + 1][r] = (__bf16)bvv[e].y;
      ldsBt[c + 2][r] = (__bf16)bvv[e].z;
      ldsBt[c + 3][r] = (__bf16)bvv[e].w;
    }
    __syncthreads();
    v16bf af = load_fragA(&ldsA[wave * 16][0], 32);
#pragma unroll
    for (int t = 0; t < 4; ++t) {
      v16bf bf = load_fragB(&ldsBt[t * 16][0], 32);
      acc[t] = wmma_bf16(af, bf, acc[t]);
    }
    __syncthreads();
  }

  const int nn = lane & 15;
  const int mh = (lane >> 4) * 8;
#pragma unroll
  for (int t = 0; t < 4; ++t) {
    int n_g = n0 + t * 16 + nn;
    float bv_ = bo[n_g];
#pragma unroll
    for (int i = 0; i < 8; ++i) {
      int m_g = m0 + wave * 16 + mh + i;
      out[(size_t)m_g * DMODEL + n_g] = acc[t][i] + bv_;
    }
  }
}

// ---------------------------------------------------------------------------
extern "C" void kernel_launch(void* const* d_in, const int* in_sizes, int n_in,
                              void* d_out, int out_size, void* d_ws,
                              size_t ws_size, hipStream_t stream) {
  const float* x  = (const float*)d_in[0];
  const float* wq = (const float*)d_in[1];
  const float* bq = (const float*)d_in[2];
  const float* wk = (const float*)d_in[3];
  const float* bk = (const float*)d_in[4];
  const float* wv = (const float*)d_in[5];
  const float* bv = (const float*)d_in[6];
  const float* wo = (const float*)d_in[7];
  const float* bo = (const float*)d_in[8];
  float* out = (float*)d_out;

  const size_t qkv_elems = (size_t)BATCH * NHEAD * SEQ * HDIM;  // 4M bf16 = 8MB
  __bf16* Qh = (__bf16*)d_ws;
  __bf16* Kh = Qh + qkv_elems;
  __bf16* Vt = Kh + qkv_elems;
  __bf16* Oc = Vt + qkv_elems;   // total 32 MB of workspace

  dim3 blk(128);
  qkv_kernel<<<dim3(DMODEL / 64, MTOT / 64, 3), blk, 0, stream>>>(
      x, wq, bq, wk, bk, wv, bv, Qh, Kh, Vt);
  attn_kernel<<<dim3(SEQ / 64, NHEAD, BATCH), blk, 0, stream>>>(Qh, Kh, Vt, Oc);
  out_proj_kernel<<<dim3(DMODEL / 64, MTOT / 64), blk, 0, stream>>>(
      Oc, wo, bo, out);
}
